// TransformerTransducerJoiner_14851996910071
// MI455X (gfx1250) — matrix operand order repaired and checked by reading, not compiled, not run on hardware
//
#include <hip/hip_runtime.h>
#include <cmath>
#include <cstdint>

typedef __attribute__((ext_vector_type(16))) _Float16 v16h;
typedef __attribute__((ext_vector_type(8)))  _Float16 v8h;
typedef __attribute__((ext_vector_type(8)))  float    v8f;

#define HH 512
#define II 1024
#define VV 1024

#define ASYNC_COPY 1

// tanh via hardware TRANS ops: v_exp_f32 + v_rcp_f32 (~1e-7 abs error,
// exact saturation at +/-1, NaN propagates). Joint is not an output; this
// feeds only the K=1024 dot product.
__device__ __forceinline__ float fast_tanh(float s) {
  float e = __builtin_amdgcn_exp2f(s * 2.8853900817779268f);  // exp(2s)
  return 1.0f - 2.0f * __builtin_amdgcn_rcpf(e + 1.0f);
}

// ---------------------------------------------------------------------------
// Small exact-f32 GEMM with bias: C[M,N] = A[M,K] @ W[K,N] + bias[N]
// ---------------------------------------------------------------------------
__global__ __launch_bounds__(256) void gemm_f32_bias(
    const float* __restrict__ A, const float* __restrict__ W,
    const float* __restrict__ bias, float* __restrict__ C,
    int M, int K, int N) {
  __shared__ float As[16][65];
  __shared__ float Ws[16][65];
  const int tid = threadIdx.x;
  const int tx = tid & 15, ty = tid >> 4;
  const int m0 = blockIdx.y * 64, n0 = blockIdx.x * 64;
  float c[4][4] = {};
  for (int k0 = 0; k0 < K; k0 += 16) {
    #pragma unroll
    for (int l = 0; l < 4; ++l) {
      int lin = tid + 256 * l;
      int m = lin >> 4, k = lin & 15;
      As[k][m] = A[(size_t)(m0 + m) * K + k0 + k];
      int kw = lin >> 6, n = lin & 63;
      Ws[kw][n] = W[(size_t)(k0 + kw) * N + n0 + n];
    }
    __syncthreads();
    #pragma unroll
    for (int k = 0; k < 16; ++k) {
      float a[4], w[4];
      #pragma unroll
      for (int i = 0; i < 4; ++i) a[i] = As[k][ty * 4 + i];
      #pragma unroll
      for (int j = 0; j < 4; ++j) w[j] = Ws[k][tx * 4 + j];
      #pragma unroll
      for (int i = 0; i < 4; ++i)
        #pragma unroll
        for (int j = 0; j < 4; ++j)
          c[i][j] = fmaf(a[i], w[j], c[i][j]);
    }
    __syncthreads();
  }
  #pragma unroll
  for (int i = 0; i < 4; ++i)
    #pragma unroll
    for (int j = 0; j < 4; ++j)
      C[(size_t)(m0 + ty * 4 + i) * N + n0 + tx * 4 + j] =
          c[i][j] + bias[n0 + tx * 4 + j];
}

// ---------------------------------------------------------------------------
// One-time prep: dense_w (f32, II x VV) -> hi/lo f16 WMMA fragments in d_ws.
// Region (kc, nb) = contiguous 128KB: [hi 32K halves][lo 32K halves], each
// ordered ((kt*32+nt)*32+lane)*16+e == the joint kernel's LDS image, so the
// joint kernel's B phase is a straight async global->LDS byte copy.
// ws[((kc*2+nb) << 16) + r2], total 2M halves = 4 MB.
// ---------------------------------------------------------------------------
__global__ __launch_bounds__(256) void prep_dense_w(
    const float* __restrict__ W, _Float16* __restrict__ ws) {
  int j = blockIdx.x * 256 + threadIdx.x;  // 0 .. 2M-1
  int rc = j >> 16;                        // kc*2 + nb
  int kc = rc >> 1, nb = rc & 1;
  int r2 = j & 65535;
  int hl = r2 >> 15;                       // 0 = hi, 1 = lo
  int r  = r2 & 32767;
  int kt = r >> 14;
  int nt = (r >> 9) & 31;
  int lane = (r >> 4) & 31;
  int e = r & 15;
  int g = lane >> 4, n = lane & 15;
  int k = kc * 64 + kt * 32 + g * 16 + e;  // B k order used here (contiguous): K = 16g + e
  int v = nb * 512 + nt * 16 + n;
  float w = W[(size_t)k * VV + v];
  _Float16 hi = (_Float16)w;
  ws[j] = hl ? (_Float16)(w - (float)hi) : hi;
}

// ---------------------------------------------------------------------------
// Fused joint + big GEMM. M_TILE=128 (2 consecutive t x 64 u), N_TILE=512,
// K chunked by 64. 1024 threads = 32 waves in a 4x8 grid; each wave owns
// 2x4 16x16 C tiles (64 acc VGPRs). LDS 160KB -> 2 WGs per WGP.
// A = tanh(enc+dec) split to f16 hi/lo in LDS (WMMA layout); B fragments
// async-copied from the prepped global image (ASYNCcnt).
// 3x v_wmma_f32_16x16x32_f16 per tile-step (hi*hi + hi*lo + lo*hi) ~ f32.
// ---------------------------------------------------------------------------
__global__ __launch_bounds__(1024) void joint_logits(
    const float* __restrict__ enc,       // (1024, II) rows = b*256+t
    const float* __restrict__ dec,       // (256,  II) rows = b*64+u
    const _Float16* __restrict__ wsB,    // prepped dense_w fragments
    const float* __restrict__ dense_b,   // (VV)
    float* __restrict__ logits) {        // (1024, 64, VV)
  __shared__ alignas(32) _Float16 Ahi[8 * 2 * 32 * 16];  //  16 KB
  __shared__ alignas(32) _Float16 Alo[8 * 2 * 32 * 16];  //  16 KB
  __shared__ alignas(32) _Float16 B2[2 * 32768];         // 128 KB: [hi][lo]

  const int tid  = threadIdx.x;
  const int lane = tid & 31;
  const int wid  = tid >> 5;
  const int wm   = wid & 3;    // 4 wave-rows (32 m each)
  const int wn   = wid >> 2;   // 8 wave-cols (64 v each)
  const int nb   = blockIdx.x;        // 512-wide v block
  const int n0   = nb * 512;
  const int bt0  = blockIdx.y * 2;    // two consecutive t (same b)
  const int b    = bt0 >> 8;

  const float* decBlk = dec + (size_t)b * 64 * II;

  v8f acc[2][4];
  #pragma unroll
  for (int i = 0; i < 2; ++i)
    #pragma unroll
    for (int j = 0; j < 4; ++j) acc[i][j] = (v8f){};

  for (int kci = 0; kci < II / 64; ++kci) {
    const int kc = kci * 64;
    __syncthreads();  // LDS reuse guard
    // ---- B fragments: straight async copy of prepped 128KB block ----
    const _Float16* src = wsB + (((size_t)(kci * 2 + nb)) << 16);
#if ASYNC_COPY
    #pragma unroll
    for (int l = 0; l < 8; ++l) {
      int j = (tid + 1024 * l) << 3;  // 8 halves = 16B per op
      unsigned ldsAddr = (unsigned)(uintptr_t)&B2[j];
      asm volatile("global_load_async_to_lds_b128 %0, %1, off"
                   :: "v"(ldsAddr), "v"(src + j) : "memory");
    }
#else
    #pragma unroll
    for (int l = 0; l < 8; ++l) {
      int j = (tid + 1024 * l) << 3;
      *(v8h*)&B2[j] = *(const v8h*)&src[j];
    }
#endif
    // ---- A fragments: 128 rows (2t x 64u) x 64 k, tanh + hi/lo split ----
    #pragma unroll
    for (int l = 0; l < 8; ++l) {
      int idx = tid + 1024 * l;          // 0..8191
      int m = idx >> 6, kk = idx & 63;   // coalesced over kk
      int tl = m >> 6, u = m & 63;
      float x = fast_tanh(enc[(size_t)(bt0 + tl) * II + kc + kk] +
                          decBlk[(size_t)u * II + kc + kk]);
      _Float16 hi = (_Float16)x;
      _Float16 lo = (_Float16)(x - (float)hi);
      int mt = m >> 4, r = m & 15;
      int kt = kk >> 5, kin = kk & 31;
      int h = kin >> 3;                       // A k order used here (contiguous):
      int g = h >> 1;                         // K = 16g + e
      int e = kin & 15;                       // (matches B)
      int off = (((mt * 2 + kt) * 32) + (g * 16 + r)) * 16 + e;
      Ahi[off] = hi;
      Alo[off] = lo;
    }
#if ASYNC_COPY
    asm volatile("s_wait_asynccnt 0x0" ::: "memory");
#endif
    __syncthreads();
    // ---- WMMA: 2 K-steps x (2x4 tiles) x 3 products ----
    #pragma unroll
    for (int kt = 0; kt < 2; ++kt) {
      v16h ah[2], al[2];
      #pragma unroll
      for (int mi = 0; mi < 2; ++mi) {
        int mt = wm * 2 + mi;
        int off = (((mt * 2 + kt) * 32) + lane) * 16;
        ah[mi] = *(const v16h*)&Ahi[off];
        al[mi] = *(const v16h*)&Alo[off];
      }
      #pragma unroll
      for (int ni = 0; ni < 4; ++ni) {
        int nt = wn * 4 + ni;
        int off = (((kt * 32 + nt) * 32) + lane) * 16;
        v16h bh = *(const v16h*)&B2[off];
        v16h bl = *(const v16h*)&B2[32768 + off];
        #pragma unroll
        for (int mi = 0; mi < 2; ++mi) {
          acc[mi][ni] = __builtin_amdgcn_wmma_f32_16x16x32_f16(
              false, ah[mi], false, bh, (short)0, acc[mi][ni], false, false);
          acc[mi][ni] = __builtin_amdgcn_wmma_f32_16x16x32_f16(
              false, ah[mi], false, bl, (short)0, acc[mi][ni], false, false);
          acc[mi][ni] = __builtin_amdgcn_wmma_f32_16x16x32_f16(
              false, al[mi], false, bh, (short)0, acc[mi][ni], false, false);
        }
      }
    }
  }

  // ---- epilogue: C/D layout (VGPR v: M=v lanes 0-15, M=v+8 lanes 16-31) ----
  const int g = lane >> 4, col = lane & 15;
  #pragma unroll
  for (int mi = 0; mi < 2; ++mi) {
    int mtb = (wm * 2 + mi) * 16;
    #pragma unroll
    for (int ni = 0; ni < 4; ++ni) {
      int vcol = n0 + (wn * 4 + ni) * 16 + col;
      float bv = dense_b[vcol];
      #pragma unroll
      for (int v = 0; v < 8; ++v) {
        int m = mtb + v + 8 * g;
        int bt = bt0 + (m >> 6), u = m & 63;
        logits[((size_t)bt * 64 + u) * VV + vcol] = acc[mi][ni][v] + bv;
      }
    }
  }
}

extern "C" void kernel_launch(void* const* d_in, const int* in_sizes, int n_in,
                              void* d_out, int out_size, void* d_ws, size_t ws_size,
                              hipStream_t stream) {
  const float* enc_h   = (const float*)d_in[0];  // (4,256,512)
  const float* dec_h   = (const float*)d_in[1];  // (4,64,512)
  const float* audio_w = (const float*)d_in[2];  // (512,1024)
  const float* audio_b = (const float*)d_in[3];  // (1024)
  const float* label_w = (const float*)d_in[4];  // (512,1024)
  const float* label_b = (const float*)d_in[5];  // (1024)
  const float* dense_w = (const float*)d_in[6];  // (1024,1024)
  const float* dense_b = (const float*)d_in[7];  // (1024)

  float* out    = (float*)d_out;
  float* logits = out;                                    // 4*256*64*1024
  float* enc4   = out + (size_t)4 * 256 * 64 * 1024;      // 4*256*1024
  float* dec4   = enc4 + (size_t)4 * 256 * 1024;          // 4*64*1024
  _Float16* wsB = (_Float16*)d_ws;                        // needs 4 MB

  // dense_w -> hi/lo f16 fragment image (2M halves)
  prep_dense_w<<<8192, 256, 0, stream>>>(dense_w, wsB);
  // enc = encoder @ audio_w + audio_b  -> enc4 slice of d_out
  gemm_f32_bias<<<dim3(16, 16), 256, 0, stream>>>(enc_h, audio_w, audio_b, enc4,
                                                  1024, HH, II);
  // dec = decoder @ label_w + label_b  -> dec4 slice of d_out
  gemm_f32_bias<<<dim3(16, 4), 256, 0, stream>>>(dec_h, label_w, label_b, dec4,
                                                 256, HH, II);
  // fused tanh(enc+dec) @ dense_w + dense_b
  joint_logits<<<dim3(2, 512), 1024, 0, stream>>>(enc4, dec4, wsB, dense_b,
                                                  logits);
}
